// MambaBlock_2963527434771
// MI455X (gfx1250) — matrix operand order repaired and checked by reading, not compiled, not run on hardware
//
#include <hip/hip_runtime.h>
#include <hip/hip_bf16.h>

// ---------------------------------------------------------------------------
// Problem constants (Mamba block)
// ---------------------------------------------------------------------------
#define BB      4
#define LL      2048
#define DMODEL  768
#define DSTATE  64
#define DCONV   4
#define DINNER  1536            // EXPAND * DMODEL
#define MROWS   (BB * LL)       // 8192

typedef __attribute__((ext_vector_type(16))) __bf16 bf16x16;
typedef __attribute__((ext_vector_type(8)))  float  f32x8;

union FragBF {
    bf16x16 v;
    uint4   q[2];
};

__device__ __forceinline__ unsigned short f32_to_bf16(float f) {
    unsigned int u = __float_as_uint(f);
    unsigned int lsb = (u >> 16) & 1u;
    u += 0x7FFFu + lsb;                      // round to nearest even
    return (unsigned short)(u >> 16);
}

// ---------------------------------------------------------------------------
// fp32 -> bf16 convert (grid-stride)
// ---------------------------------------------------------------------------
__global__ void cvt_f32_bf16(const float* __restrict__ src,
                             unsigned short* __restrict__ dst, int n) {
    int i = blockIdx.x * blockDim.x + threadIdx.x;
    int stride = gridDim.x * blockDim.x;
    for (; i < n; i += stride) dst[i] = f32_to_bf16(src[i]);
}

// ---------------------------------------------------------------------------
// WMMA GEMM:  C[M,N] = A[M,K] * W[N,K]^T + bias
//   A, W bf16 (K-contiguous rows), fp32 accumulate.
//   Block: 256 threads = 8 waves (4 M-waves x 2 N-waves),
//   block tile 128x128, wave tile 32x64 (2x4 WMMA 16x16x32 tiles):
//   12 b128 loads per 8 WMMAs per K-slab (1.5 loads/WMMA).
// Epilogue MODE:
//   0: plain fp32 store to out0 (ld = Ndim)                       [out_proj]
//   1: col<1536 -> out0 (xi, ld 1536); else out1 (res, ld 1536)   [in_proj]
//   2: col<1536 -> outb bf16 (delta_in, ld 1536);
//      col<1600 -> out0 (Bm, ld 64); else out1 (Cm, ld 64)        [x_proj]
//   3: softplus -> out0 fp32 (ld Ndim)                            [dt_proj]
// ---------------------------------------------------------------------------
template <int MODE>
__global__ __launch_bounds__(256)
void gemm_wmma_bf16(const unsigned short* __restrict__ Abf,
                    const unsigned short* __restrict__ Wbf,
                    const float* __restrict__ bias,
                    int Ndim, int K,
                    float* __restrict__ out0,
                    float* __restrict__ out1,
                    unsigned short* __restrict__ outb) {
    const int lane  = threadIdx.x & 31;
    const int wave  = threadIdx.x >> 5;      // 0..7
    const int wrow  = wave >> 1;             // 0..3 -> M
    const int wcol  = wave & 1;              // 0..1 -> N
    const int M0    = blockIdx.x * 128 + wrow * 32;
    const int N0    = blockIdx.y * 128 + wcol * 64;
    const int lhalf = lane >> 4;             // 0/1
    const int l16   = lane & 15;

    f32x8 acc[2][4] = {};

    // A fragment (16x32, 16-bit): lane holds row M0+l16 (+16 for second
    // tile), K-chunks {kk + 8*lhalf .. +7} and {kk + 8*lhalf + 16 .. +7}
    const unsigned short* aptr[2];
    aptr[0] = Abf + (size_t)(M0 + l16) * K + lhalf * 8;
    aptr[1] = aptr[0] + (size_t)16 * K;
    // B fragment (32x16): lane l16 holds column N0+tj*16+l16 = row of W,
    // K-chunks {kk + 8*lhalf .. +7} and {kk + 8*lhalf + 16 .. +7}
    const unsigned short* bptr[4];
    #pragma unroll
    for (int j = 0; j < 4; ++j)
        bptr[j] = Wbf + (size_t)(N0 + j * 16 + l16) * K + lhalf * 8;

    for (int kk = 0; kk < K; kk += 32) {
        FragBF fa[2], fb[4];
        #pragma unroll
        for (int i = 0; i < 2; ++i) {
            fa[i].q[0] = *(const uint4*)(aptr[i] + kk);
            fa[i].q[1] = *(const uint4*)(aptr[i] + kk + 16);
        }
        #pragma unroll
        for (int j = 0; j < 4; ++j) {
            fb[j].q[0] = *(const uint4*)(bptr[j] + kk);
            fb[j].q[1] = *(const uint4*)(bptr[j] + kk + 16);
        }

        if (kk + 32 < K) {   // next-slab prefetch into near caches
            __builtin_prefetch(aptr[0] + kk + 32, 0, 3);
            __builtin_prefetch(aptr[1] + kk + 32, 0, 3);
            __builtin_prefetch(bptr[0] + kk + 32, 0, 3);
            __builtin_prefetch(bptr[2] + kk + 32, 0, 3);
        }

        #pragma unroll
        for (int i = 0; i < 2; ++i) {
            #pragma unroll
            for (int j = 0; j < 4; ++j) {
                acc[i][j] = __builtin_amdgcn_wmma_f32_16x16x32_bf16(
                    false, fa[i].v, false, fb[j].v, (short)0, acc[i][j],
                    false, false);
            }
        }
    }

    // Epilogue. C/D layout: lane l16 -> column, VGPR r -> row r + 8*lhalf.
    #pragma unroll
    for (int ti = 0; ti < 2; ++ti) {
        #pragma unroll
        for (int tj = 0; tj < 4; ++tj) {
            const int col = N0 + tj * 16 + l16;
            const float bv = bias[col];
            #pragma unroll
            for (int r = 0; r < 8; ++r) {
                const int row = M0 + ti * 16 + r + 8 * lhalf;
                float v = acc[ti][tj][r] + bv;
                if (MODE == 0) {
                    out0[(size_t)row * Ndim + col] = v;
                } else if (MODE == 1) {
                    if (col < DINNER) out0[(size_t)row * DINNER + col] = v;
                    else              out1[(size_t)row * DINNER + (col - DINNER)] = v;
                } else if (MODE == 2) {
                    if (col < DINNER)
                        outb[(size_t)row * DINNER + col] = f32_to_bf16(v);
                    else if (col < DINNER + DSTATE)
                        out0[(size_t)row * DSTATE + (col - DINNER)] = v;
                    else
                        out1[(size_t)row * DSTATE + (col - DINNER - DSTATE)] = v;
                } else { // MODE 3: softplus
                    float sp = (v > 20.f) ? v : log1pf(__expf(v));
                    out0[(size_t)row * Ndim + col] = sp;
                }
            }
        }
    }
}

// ---------------------------------------------------------------------------
// Depthwise causal conv (width 4) + SiLU.  xi layout: [(b*L+l), d]
// ---------------------------------------------------------------------------
__global__ void conv_silu_kernel(const float* __restrict__ xi,
                                 const float* __restrict__ conv_w,
                                 const float* __restrict__ conv_b,
                                 float* __restrict__ u_f32,
                                 unsigned short* __restrict__ u_bf) {
    const int idx = blockIdx.x * blockDim.x + threadIdx.x;
    if (idx >= BB * LL * DINNER) return;
    const int d = idx % DINNER;
    const int l = (idx / DINNER) % LL;
    const int b = idx / (DINNER * LL);

    const float* base = xi + ((size_t)b * LL) * DINNER + d;
    float acc = conv_b[d];
    #pragma unroll
    for (int k = 0; k < DCONV; ++k) {
        const int ll = l - (DCONV - 1) + k;
        if (ll >= 0) acc += base[(size_t)ll * DINNER] * conv_w[d * DCONV + k];
    }
    const float s = acc / (1.f + __expf(-acc));   // silu
    u_f32[idx] = s;
    u_bf[idx]  = f32_to_bf16(s);
}

// ---------------------------------------------------------------------------
// Selective scan. One lane per (b, d) channel; 64-state + A-row in VGPRs.
// B_t / C_t staged through LDS once per timestep (shared by all channels).
// Epilogue fused: y = (y + D*u) * silu(res), stored bf16 for out_proj GEMM.
// grid = BB * (DINNER/256), block = 256.
// ---------------------------------------------------------------------------
__global__ __launch_bounds__(256)
void scan_kernel(const float* __restrict__ delta,
                 const float* __restrict__ u_f32,
                 const float* __restrict__ Bm,
                 const float* __restrict__ Cm,
                 const float* __restrict__ A_param,
                 const float* __restrict__ D_param,
                 const float* __restrict__ res,
                 unsigned short* __restrict__ y_bf) {
    const int tid = threadIdx.x;
    const int chunks = DINNER / 256;                   // 6
    const int b = blockIdx.x / chunks;
    const int d = (blockIdx.x % chunks) * 256 + tid;

    __shared__ float sB[DSTATE];
    __shared__ float sC[DSTATE];

    float A[DSTATE], s[DSTATE];
    #pragma unroll
    for (int n = 0; n < DSTATE; ++n) {
        A[n] = -__expf(A_param[(size_t)d * DSTATE + n]);
        s[n] = 0.f;
    }
    const float Dv = D_param[d];

    for (int l = 0; l < LL; ++l) {
        const size_t row = (size_t)b * LL + l;
        if (tid < DSTATE)            sB[tid]          = Bm[row * DSTATE + tid];
        else if (tid < 2 * DSTATE)   sC[tid - DSTATE] = Cm[row * DSTATE + (tid - DSTATE)];
        __syncthreads();

        const float dv = delta[row * DINNER + d];
        const float uv = u_f32[row * DINNER + d];
        const float du = dv * uv;
        float y = 0.f;
        #pragma unroll
        for (int n = 0; n < DSTATE; ++n) {
            s[n] = __expf(dv * A[n]) * s[n] + du * sB[n];
            y += s[n] * sC[n];
        }

        const float r  = res[row * DINNER + d];
        const float yv = (y + Dv * uv) * (r / (1.f + __expf(-r)));
        y_bf[row * DINNER + d] = f32_to_bf16(yv);
        __syncthreads();
    }
}

// ---------------------------------------------------------------------------
// Workspace layout (bytes, all 256B aligned)
// ---------------------------------------------------------------------------
#define OFF_WB_IN   ((size_t)0)                                  // 3072*768 bf16
#define OFF_WB_X    (OFF_WB_IN  + (size_t)3072 * 768 * 2)        // 1664*1536 bf16
#define OFF_WB_DT   (OFF_WB_X   + (size_t)1664 * 1536 * 2)       // 1536*1536 bf16
#define OFF_WB_OUT  (OFF_WB_DT  + (size_t)1536 * 1536 * 2)       // 768*1536 bf16
#define OFF_XBF     (OFF_WB_OUT + (size_t)768 * 1536 * 2)        // M*768 bf16
#define OFF_XI      (OFF_XBF    + (size_t)MROWS * DMODEL * 2)    // M*1536 f32
#define OFF_RES     (OFF_XI     + (size_t)MROWS * DINNER * 4)    // M*1536 f32
#define OFF_UF      (OFF_RES    + (size_t)MROWS * DINNER * 4)    // M*1536 f32
#define OFF_UBF     (OFF_UF     + (size_t)MROWS * DINNER * 4)    // M*1536 bf16
#define OFF_DINBF   (OFF_UBF    + (size_t)MROWS * DINNER * 2)    // M*1536 bf16
#define OFF_BM      (OFF_DINBF  + (size_t)MROWS * DINNER * 2)    // M*64 f32
#define OFF_CM      (OFF_BM     + (size_t)MROWS * DSTATE * 4)    // M*64 f32
#define OFF_DELTA   (OFF_CM     + (size_t)MROWS * DSTATE * 4)    // M*1536 f32
#define OFF_YBF     (OFF_DELTA  + (size_t)MROWS * DINNER * 4)    // M*1536 bf16

extern "C" void kernel_launch(void* const* d_in, const int* in_sizes, int n_in,
                              void* d_out, int out_size, void* d_ws, size_t ws_size,
                              hipStream_t stream) {
    const float* x          = (const float*)d_in[0];
    const float* in_proj_w  = (const float*)d_in[1];
    const float* in_proj_b  = (const float*)d_in[2];
    const float* conv_w     = (const float*)d_in[3];
    const float* conv_b     = (const float*)d_in[4];
    const float* x_proj_w   = (const float*)d_in[5];
    const float* x_proj_b   = (const float*)d_in[6];
    const float* dt_proj_w  = (const float*)d_in[7];
    const float* dt_proj_b  = (const float*)d_in[8];
    const float* A_param    = (const float*)d_in[9];
    const float* D_param    = (const float*)d_in[10];
    const float* out_proj_w = (const float*)d_in[11];
    const float* out_proj_b = (const float*)d_in[12];
    float* out = (float*)d_out;

    char* ws = (char*)d_ws;
    unsigned short* wb_in  = (unsigned short*)(ws + OFF_WB_IN);
    unsigned short* wb_x   = (unsigned short*)(ws + OFF_WB_X);
    unsigned short* wb_dt  = (unsigned short*)(ws + OFF_WB_DT);
    unsigned short* wb_out = (unsigned short*)(ws + OFF_WB_OUT);
    unsigned short* x_bf   = (unsigned short*)(ws + OFF_XBF);
    float*          xi     = (float*)(ws + OFF_XI);
    float*          res    = (float*)(ws + OFF_RES);
    float*          u_f32  = (float*)(ws + OFF_UF);
    unsigned short* u_bf   = (unsigned short*)(ws + OFF_UBF);
    unsigned short* din_bf = (unsigned short*)(ws + OFF_DINBF);
    float*          Bm     = (float*)(ws + OFF_BM);
    float*          Cm     = (float*)(ws + OFF_CM);
    float*          delta  = (float*)(ws + OFF_DELTA);
    unsigned short* y_bf   = (unsigned short*)(ws + OFF_YBF);

    // --- fp32 -> bf16 conversions (weights + x) ---
    auto cvt = [&](const float* s, unsigned short* d, int n) {
        cvt_f32_bf16<<<(n + 255) / 256, 256, 0, stream>>>(s, d, n);
    };
    cvt(in_proj_w,  wb_in,  2 * DINNER * DMODEL);          // 3072*768
    cvt(x_proj_w,   wb_x,   (DINNER + 2 * DSTATE) * DINNER);
    cvt(dt_proj_w,  wb_dt,  DINNER * DINNER);
    cvt(out_proj_w, wb_out, DMODEL * DINNER);
    cvt(x,          x_bf,   MROWS * DMODEL);

    // --- GEMM 1: xr = x @ in_proj_w^T + b  ->  xi | res ---
    {
        dim3 grid(MROWS / 128, (2 * DINNER) / 128);
        gemm_wmma_bf16<1><<<grid, 256, 0, stream>>>(
            x_bf, wb_in, in_proj_b, 2 * DINNER, DMODEL, xi, res, nullptr);
    }

    // --- depthwise conv + silu -> u ---
    {
        const int n = BB * LL * DINNER;
        conv_silu_kernel<<<(n + 255) / 256, 256, 0, stream>>>(
            xi, conv_w, conv_b, u_f32, u_bf);
    }

    // --- GEMM 2: dbc = u @ x_proj_w^T + b -> delta_in(bf16) | Bm | Cm ---
    {
        dim3 grid(MROWS / 128, (DINNER + 2 * DSTATE) / 128);
        gemm_wmma_bf16<2><<<grid, 256, 0, stream>>>(
            u_bf, wb_x, x_proj_b, DINNER + 2 * DSTATE, DINNER, Bm, Cm, din_bf);
    }

    // --- GEMM 3: delta = softplus(delta_in @ dt_proj_w^T + b) ---
    {
        dim3 grid(MROWS / 128, DINNER / 128);
        gemm_wmma_bf16<3><<<grid, 256, 0, stream>>>(
            din_bf, wb_dt, dt_proj_b, DINNER, DINNER, delta, nullptr, nullptr);
    }

    // --- selective scan (+ fused D*u and silu(res) gating) -> y(bf16) ---
    {
        dim3 grid(BB * (DINNER / 256));
        scan_kernel<<<grid, 256, 0, stream>>>(
            delta, u_f32, Bm, Cm, A_param, D_param, res, y_bf);
    }

    // --- GEMM 4: out = y @ out_proj_w^T + b ---
    {
        dim3 grid(MROWS / 128, DMODEL / 128);
        gemm_wmma_bf16<0><<<grid, 256, 0, stream>>>(
            y_bf, wb_out, out_proj_b, DMODEL, DINNER, out, nullptr, nullptr);
    }
}
